// BasicModel_43619687858602
// MI455X (gfx1250) — matrix operand order repaired and checked by reading, not compiled, not run on hardware
//
#include <hip/hip_runtime.h>

typedef __attribute__((ext_vector_type(16))) _Float16 v16h;
typedef __attribute__((ext_vector_type(8)))  _Float16 v8h;
typedef __attribute__((ext_vector_type(4)))  _Float16 v4h;
typedef __attribute__((ext_vector_type(8)))  float    v8f;

union F16Frag { v16h v; v8h h8[2]; };

// Problem constants (from reference)
constexpr int BB    = 16;
constexpr int NPTS  = 16384;
constexpr int FD    = 192;
constexpr int GC    = 63;             // grid rows == cols
constexpr int SEG   = GC * GC;        // 3969 cells per batch
constexpr int ROWS  = 16 * 63 * 62;   // 62496 rows per head
constexpr int TILE_M = 48;            // 48 | 62496 -> no ragged tiles
constexpr int NTILES = ROWS / TILE_M; // 1302
constexpr int HEAD_WT = 141824;       // padded f16 weight elements per head

// device-side copies of layer tables (for prep kernel)
__device__ __constant__ const int KDIM[5] = {384, 256, 128, 64, 32};
__device__ __constant__ const int NACTC[5] = {256, 128, 64, 32, 2};
__device__ __constant__ const int WOFF[6] = {0, 98304, 131072, 139264, 141312, 141824};

// ws layout
constexpr size_t POOLED_ELEMS = (size_t)BB * SEG * FD;             // 12,192,768
constexpr size_t POOLED_BYTES = POOLED_ELEMS * 4;                  // 48,771,072
constexpr size_t COUNT_ELEMS  = (size_t)BB * SEG;                  // 63,504
constexpr size_t COUNT_BYTES  = COUNT_ELEMS * 4;                   // 254,016
constexpr size_t WT_OFF       = POOLED_BYTES + COUNT_BYTES;        // 16B aligned

// ---- order-preserving float<->uint encoding for atomic max ----
__device__ __forceinline__ unsigned encf(float f) {
    unsigned u = __float_as_uint(f);
    return (u & 0x80000000u) ? ~u : (u | 0x80000000u);
}
__device__ __forceinline__ float decf(unsigned u) {
    return __uint_as_float((u & 0x80000000u) ? (u ^ 0x80000000u) : ~u);
}

// K1: zero pooled-encoded buffer + counts (contiguous in ws)
__global__ __launch_bounds__(256) void k_init(unsigned* __restrict__ p, size_t n) {
    size_t i = (size_t)blockIdx.x * 256 + threadIdx.x;
    if (i < n) p[i] = 0u;
}

// K2: fused BatchNorm(eval) + segment-max scatter (+ occupancy counts)
__global__ __launch_bounds__(256) void k_bn_scatter(
    const float* __restrict__ gf, const int* __restrict__ cid,
    const float* __restrict__ gamma, const float* __restrict__ beta,
    const float* __restrict__ mean, const float* __restrict__ var,
    unsigned* __restrict__ pooledU, int* __restrict__ counts) {
    size_t i = (size_t)blockIdx.x * 256 + threadIdx.x;
    if (i >= (size_t)BB * NPTS * FD) return;
    int f = (int)(i % FD);
    size_t v = i / FD;                  // global vertex
    int b = (int)(v / NPTS);
    int r = cid[2 * v];
    int c = cid[2 * v + 1];
    int seg = r * GC + c;
    float x = gf[i];
    float val = gamma[f] * (x - mean[f]) * rsqrtf(var[f] + 1e-5f) + beta[f];
    atomicMax(pooledU + ((size_t)b * SEG + seg) * FD + f, encf(val));
    if (f == 0) atomicAdd(counts + b * SEG + seg, 1);
}

// K3: decode in place; empty cells -> 0
__global__ __launch_bounds__(256) void k_finalize(unsigned* __restrict__ pooledU,
                                                  const int* __restrict__ counts) {
    size_t i = (size_t)blockIdx.x * 256 + threadIdx.x;
    if (i >= POOLED_ELEMS) return;
    size_t cell = i / FD;
    float o = (counts[cell] > 0) ? decf(pooledU[i]) : 0.0f;
    ((float*)pooledU)[i] = o;
}

// K4: weights f32 [K][Nact] -> f16 transposed [Npad][K] (B-operand layout), padded cols = 0
__global__ __launch_bounds__(256) void k_prep_weights(
    const float* w00, const float* w01, const float* w02, const float* w03, const float* w04,
    const float* w10, const float* w11, const float* w12, const float* w13, const float* w14,
    _Float16* __restrict__ wt) {
    const float* Ws[10] = {w00, w01, w02, w03, w04, w10, w11, w12, w13, w14};
    int idx = blockIdx.x * 256 + threadIdx.x;
    if (idx >= 2 * HEAD_WT) return;
    int head = idx / HEAD_WT;
    int rem  = idx % HEAD_WT;
    int l = 0;
    while (rem >= WOFF[l + 1]) ++l;
    int loc = rem - WOFF[l];
    int K = KDIM[l], Na = NACTC[l];
    int n = loc / K, k = loc % K;
    float v = (n < Na) ? Ws[head * 5 + l][k * Na + n] : 0.0f;
    wt[idx] = (_Float16)v;
}

// ---- One MLP layer as a compile-time-shaped WMMA GEMM ----
// nt-major wave loop: one B fragment load feeds MT=3 WMMAs (B reuse, latency hiding)
template<int K, int Na, int Np, bool LAST>
__device__ __forceinline__ void layer_gemm(
    const _Float16* __restrict__ Acur, _Float16* __restrict__ Anext,
    const _Float16* __restrict__ W, const float* __restrict__ bptr,
    float* __restrict__ out, int m0, int head, int tid) {
    const int lane = tid & 31;
    const int wv   = tid >> 5;
    const int ln   = lane & 15;
    const int kloA = (lane < 16) ? 0 : 8;    // A-operand K split {0-7,16-23}/{8-15,24-31}
    const int kloB = (lane < 16) ? 0 : 8;    // B-operand K split {0-7,16-23}/{8-15,24-31}
    const int hiM  = (lane < 16) ? 0 : 8;    // C/D row offset for upper half-wave
    constexpr int NT = Np / 16;
    constexpr int MT = TILE_M / 16;

    for (int nt = wv; nt < NT; nt += 8) {
        const int col = nt * 16 + ln;
        const float bv = (col < Na) ? bptr[col] : 0.0f;
        v8f acc[MT];
#pragma unroll
        for (int mt = 0; mt < MT; ++mt)
#pragma unroll
            for (int vi = 0; vi < 8; ++vi) acc[mt][vi] = bv;

        const _Float16* bcol = W + (size_t)col * K + kloB;
        __builtin_prefetch(bcol, 0, 1);      // global_prefetch_b8

#pragma unroll 4
        for (int k0 = 0; k0 < K; k0 += 32) {
            F16Frag bf;
            bf.h8[0] = *(const v8h*)(bcol + k0);          // global b128 (L2-hot)
            bf.h8[1] = *(const v8h*)(bcol + k0 + 16);
#pragma unroll
            for (int mt = 0; mt < MT; ++mt) {
                const _Float16* arow = Acur + (size_t)(mt * 16 + ln) * K + kloA + k0;
                F16Frag a;
                a.h8[0] = *(const v8h*)(arow);            // LDS b128
                a.h8[1] = *(const v8h*)(arow + 16);
                acc[mt] = __builtin_amdgcn_wmma_f32_16x16x32_f16(
                    false, a.v, false, bf.v, (short)0, acc[mt], false, false);
            }
        }

#pragma unroll
        for (int mt = 0; mt < MT; ++mt) {
            if (!LAST) {
#pragma unroll
                for (int vi = 0; vi < 8; ++vi) {
                    float x = acc[mt][vi];
                    x = x > 0.0f ? x : 0.01f * x;         // LeakyReLU
                    int row = mt * 16 + vi + hiM;
                    Anext[(size_t)row * Na + col] = (_Float16)x;
                }
            } else if (col < 2) {
#pragma unroll
                for (int vi = 0; vi < 8; ++vi) {
                    float x = acc[mt][vi];
                    x = x > 0.0f ? x : 0.0f;              // final ReLU
                    int row = mt * 16 + vi + hiM;
                    size_t R = (size_t)m0 + row;
                    out[(size_t)head * ROWS * 2 + R * 2 + col] = x;
                }
            }
        }
    }
    __syncthreads();
}

// K5: fully fused 5-layer MLP per head, WMMA f32<-f16
__global__ __launch_bounds__(256) void k_mlp(
    const float* __restrict__ pooled, const _Float16* __restrict__ wt,
    const float* b00, const float* b01, const float* b02, const float* b03, const float* b04,
    const float* b10, const float* b11, const float* b12, const float* b13, const float* b14,
    float* __restrict__ out) {
    __shared__ _Float16 bufA[TILE_M * 384];   // 36,864 B
    __shared__ _Float16 bufB[TILE_M * 256];   // 24,576 B  (total 61,440 B < 64 KB static)

    const float* biases[10] = {b00, b01, b02, b03, b04, b10, b11, b12, b13, b14};

    const int head = blockIdx.y;
    const int m0   = blockIdx.x * TILE_M;
    const int tid  = threadIdx.x;

    // ---- Stage 0: gather right/down rows from pooled (float4), cvt -> f16 LDS ----
    for (int idx = tid; idx < TILE_M * 96; idx += 256) {
        int r = idx / 96, c = (idx % 96) * 4;
        int R = m0 + r;
        float4 v;
        if (head == 0) {  // right: cells (i,j),(i,j+1) are 384 contiguous floats
            int b = R / (63 * 62); int rem = R % (63 * 62);
            int i = rem / 62, j = rem % 62;
            v = *(const float4*)(pooled + (((size_t)b * 63 + i) * 63 + j) * FD + c);
        } else {          // down: cells (i,j),(i+1,j)
            int b = R / (62 * 63); int rem = R % (62 * 63);
            int i = rem / 63, j = rem % 63;
            size_t base = (((size_t)b * 63 + i) * 63 + j) * FD;
            const float* src = (c < FD) ? (pooled + base + c)
                                        : (pooled + base + (size_t)63 * FD + (c - FD));
            v = *(const float4*)src;
        }
        v4h h;
        h[0] = (_Float16)v.x; h[1] = (_Float16)v.y;
        h[2] = (_Float16)v.z; h[3] = (_Float16)v.w;
        *(v4h*)(bufA + (size_t)r * 384 + c) = h;
    }
    __syncthreads();

    const _Float16* Wh = wt + (size_t)head * HEAD_WT;
    const float* const* bh = biases + head * 5;

    layer_gemm<384, 256, 256, false>(bufA, bufB, Wh + 0,      bh[0], out, m0, head, tid);
    layer_gemm<256, 128, 128, false>(bufB, bufA, Wh + 98304,  bh[1], out, m0, head, tid);
    layer_gemm<128,  64,  64, false>(bufA, bufB, Wh + 131072, bh[2], out, m0, head, tid);
    layer_gemm< 64,  32,  32, false>(bufB, bufA, Wh + 139264, bh[3], out, m0, head, tid);
    layer_gemm< 32,   2,  16, true >(bufA, bufB, Wh + 141312, bh[4], out, m0, head, tid);
}

extern "C" void kernel_launch(void* const* d_in, const int* in_sizes, int n_in,
                              void* d_out, int out_size, void* d_ws, size_t ws_size,
                              hipStream_t stream) {
    const float* gf    = (const float*)d_in[0];
    const int*   cid   = (const int*)d_in[1];
    const float* gamma = (const float*)d_in[2];
    const float* beta  = (const float*)d_in[3];
    const float* mean  = (const float*)d_in[4];
    const float* var   = (const float*)d_in[5];
    // head0: d_in[6..15] = W0,b0,...,W4,b4 ; head1: d_in[16..25]
    const float* W0[5] = {(const float*)d_in[6],  (const float*)d_in[8],
                          (const float*)d_in[10], (const float*)d_in[12],
                          (const float*)d_in[14]};
    const float* B0[5] = {(const float*)d_in[7],  (const float*)d_in[9],
                          (const float*)d_in[11], (const float*)d_in[13],
                          (const float*)d_in[15]};
    const float* W1[5] = {(const float*)d_in[16], (const float*)d_in[18],
                          (const float*)d_in[20], (const float*)d_in[22],
                          (const float*)d_in[24]};
    const float* B1[5] = {(const float*)d_in[17], (const float*)d_in[19],
                          (const float*)d_in[21], (const float*)d_in[23],
                          (const float*)d_in[25]};

    unsigned*  pooledU = (unsigned*)d_ws;
    int*       counts  = (int*)((char*)d_ws + POOLED_BYTES);
    _Float16*  wt      = (_Float16*)((char*)d_ws + WT_OFF);

    // K1: zero pooled (encoded -inf == 0) and counts (contiguous)
    {
        size_t n = POOLED_ELEMS + COUNT_ELEMS;
        k_init<<<dim3((unsigned)((n + 255) / 256)), 256, 0, stream>>>(pooledU, n);
    }
    // K2: fused BN + atomic segment-max scatter
    {
        size_t n = (size_t)BB * NPTS * FD;  // 50,331,648
        k_bn_scatter<<<dim3((unsigned)(n / 256)), 256, 0, stream>>>(
            gf, cid, gamma, beta, mean, var, pooledU, counts);
    }
    // K3: decode + zero empty cells
    k_finalize<<<dim3((unsigned)(POOLED_ELEMS / 256)), 256, 0, stream>>>(pooledU, counts);
    // K4: weights -> f16 transposed/padded (B-operand layout)
    k_prep_weights<<<dim3((2 * HEAD_WT) / 256), 256, 0, stream>>>(
        W0[0], W0[1], W0[2], W0[3], W0[4], W1[0], W1[1], W1[2], W1[3], W1[4], wt);
    // K5: fused MLP heads (grid.y = head)
    k_mlp<<<dim3(NTILES, 2), 256, 0, stream>>>(
        (const float*)pooledU, wt,
        B0[0], B0[1], B0[2], B0[3], B0[4], B1[0], B1[1], B1[2], B1[3], B1[14 - 14 + 4],
        (float*)d_out);
}
